// DMSAttention_10222022164489
// MI455X (gfx1250) — matrix operand order repaired and checked by reading, not compiled, not run on hardware
//
#include <hip/hip_runtime.h>
#include <hip/hip_bf16.h>
#include <math.h>

// ---------------- problem constants ----------------
#define BB     8
#define CCH    384
#define HH     14
#define WWD    14
#define NNT    196          // H*W tokens
#define SSD    49           // rH*rW deformable samples
#define KKN    9            // 3x3 center samples
#define STT    58           // SSD + KKN
#define NHEAD  12
#define HCD    32
#define RH     7
#define TBD    27           // rel-pos table 2H-1

typedef __attribute__((ext_vector_type(16))) __bf16 v16bf;
typedef __attribute__((ext_vector_type(8)))  float  v8f;
typedef int v4i __attribute__((vector_size(16)));   // matches async-lds builtin param type

// fragment <-> two 16B LDS loads
typedef union { v16bf v; uint4 q[2]; } fragU;
// 8 bf16 <-> one 16B global load
typedef union { uint4 q; __bf16 h[8]; } W8;

// ---- gfx1250 async global->LDS path (guarded; falls back to sync copies) ----
#if defined(__has_builtin)
#  if __has_builtin(__builtin_amdgcn_global_load_async_to_lds_b128)
#    define HAVE_ASYNC_LDS 1
#  endif
#endif

#ifdef HAVE_ASYNC_LDS
#  define ASYNC_CP16(dst, src) \
     __builtin_amdgcn_global_load_async_to_lds_b128( \
         reinterpret_cast<v4i*>(const_cast<__bf16*>(src)), \
         reinterpret_cast<v4i*>(dst), 0, 0)
#  if __has_builtin(__builtin_amdgcn_s_wait_asynccnt)
#    define WAIT_ASYNC(n) __builtin_amdgcn_s_wait_asynccnt(n)
#  else
#    define WAIT_ASYNC(n) asm volatile("s_wait_asynccnt %0" :: "i"(n) : "memory")
#  endif
#else
#  define ASYNC_CP16(dst, src) \
     (*reinterpret_cast<uint4*>(dst) = *reinterpret_cast<const uint4*>(src))
#  define WAIT_ASYNC(n) ((void)0)
#endif

__device__ __constant__ float OM[9][2] = {
  {0.f,-1.f},{-1.f,-1.f},{-1.f,0.f},{-1.f,1.f},{0.f,1.f},
  {1.f,1.f},{1.f,0.f},{1.f,-1.f},{0.f,0.f}};

__device__ inline float gelu_f(float x){
  return 0.5f * x * (1.0f + erff(x * 0.70710678118654752f));
}

// bilinear_sample weights/indices, grid_sample align_corners=True, zeros padding
__device__ inline void bili(float cy, float cx, int Hi, int Wi, float* w, int* id){
  float py = (cy + 1.f) * 0.5f * (float)(Hi - 1);
  float px = (cx + 1.f) * 0.5f * (float)(Wi - 1);
  float y0 = floorf(py), x0 = floorf(px);
  float wy = py - y0, wx = px - x0;
  float ys[4] = {y0, y0, y0 + 1.f, y0 + 1.f};
  float xs[4] = {x0, x0 + 1.f, x0, x0 + 1.f};
  float ww[4] = {(1.f-wy)*(1.f-wx), (1.f-wy)*wx, wy*(1.f-wx), wy*wx};
#pragma unroll
  for (int j = 0; j < 4; ++j){
    bool v = (ys[j] >= 0.f) && (ys[j] <= (float)(Hi-1)) &&
             (xs[j] >= 0.f) && (xs[j] <= (float)(Wi-1));
    int yi = (int)fminf(fmaxf(ys[j], 0.f), (float)(Hi-1));
    int xi = (int)fminf(fmaxf(xs[j], 0.f), (float)(Wi-1));
    w[j]  = v ? ww[j] : 0.f;
    id[j] = yi * Wi + xi;
  }
}

// ---------------- K0: NCHW -> NHWC (f32 + bf16 copy) ----------------
__global__ void k_xpose(const float* __restrict__ x, float* __restrict__ xh,
                        __bf16* __restrict__ xbf){
  int idx = blockIdx.x * blockDim.x + threadIdx.x;
  if (idx >= BB * NNT * CCH) return;
  int c = idx % CCH;
  int n = (idx / CCH) % NNT;
  int b = idx / (CCH * NNT);
  float v = x[(b * CCH + c) * NNT + n];
  xh[idx]  = v;
  xbf[idx] = (__bf16)v;
}

// ---------------- K1a: plain f32 -> bf16 convert (same row-major layout) ----------------
__global__ void k_cvt(const float* __restrict__ Wsrc, __bf16* __restrict__ Wbf, int nel){
  int idx = blockIdx.x * blockDim.x + threadIdx.x;
  if (idx < nel) Wbf[idx] = (__bf16)Wsrc[idx];
}

// ---------------- K1b: transpose convert  WT[c*OC+o] = W[o*IC+c]  (for q->Wk proj) ----
__global__ void k_wt(const float* __restrict__ Wsrc, __bf16* __restrict__ WT,
                     int IC, int OC){
  int idx = blockIdx.x * blockDim.x + threadIdx.x;
  if (idx >= IC * OC) return;
  int o = idx % OC;
  int c = idx / OC;
  WT[idx] = (__bf16)Wsrc[o * IC + c];
}

// ---------------- K2: WMMA bf16 GEMM  D = A(MxK) * W^T + bias, W row-major [N][K] ------
// flags: bit0 = GELU epilogue ; bit1 = NCHW scatter (M = B*196, N = 384)
// Kd must be a multiple of 32 (it is always 384 here).
// Double-buffered LDS tiles filled with async global->LDS copies (ASYNCcnt).
__global__ __launch_bounds__(256) void k_gemm(
    const __bf16* __restrict__ A, const __bf16* __restrict__ Wn,
    const float* __restrict__ bias, float* __restrict__ outF,
    __bf16* __restrict__ outB, int M, int N, int Kd, int flags)
{
  __shared__ __align__(16) __bf16 sA [2][64 * 32];   // [buf][m][k]
  __shared__ __align__(16) __bf16 sBn[2][32 * 32];   // [buf][n][k] == W rows (B-frag major)
  const int tid  = threadIdx.x;
  const int m0   = blockIdx.y * 64;
  const int n0   = blockIdx.x * 32;
  const int wv   = tid >> 5, lane = tid & 31;
  const int mt   = (wv & 3) * 16, nt = (wv >> 2) * 16;
  const int lm   = lane & 15, lh = lane >> 4;

  const int baseA = (mt + lm) * 32 + 8 * lh;      // A frag: k = 8h..8h+7, 16+8h..+23
  const int baseB = (nt + lm) * 32 + 8 * lh;      // B frag: k = 8h..8h+7, 16+8h..+23

  // per-thread staging slots (one b128 each)
  const int ar = tid >> 2, aq = tid & 3;          // A: 64 rows x 4 quads (all 256 threads)
  const int br = (tid & 127) >> 2, bq2 = tid & 3; // B: 32 rows x 4 quads (threads 0..127)
  const int am  = m0 + ar;
  const int bn2 = n0 + br;
  const int sltA = ar * 32 + aq * 8;
  const int sltB = br * 32 + bq2 * 8;
  const bool doA = (am < M);
  const bool doB = (tid < 128) && (bn2 < N);
  const bool anyA = __any((int)doA) != 0;                       // wave-uniform
  const bool anyB = (tid < 128) && (__any((int)(bn2 < N)) != 0);// wave-uniform

  // zero-fill out-of-bounds slots once, in both buffers (never async-written)
  if (!doA){
    uint4 z = make_uint4(0u,0u,0u,0u);
    *reinterpret_cast<uint4*>(&sA[0][sltA]) = z;
    *reinterpret_cast<uint4*>(&sA[1][sltA]) = z;
  }
  if (tid < 128 && bn2 >= N){
    uint4 z = make_uint4(0u,0u,0u,0u);
    *reinterpret_cast<uint4*>(&sBn[0][sltB]) = z;
    *reinterpret_cast<uint4*>(&sBn[1][sltB]) = z;
  }

  const int nT = Kd >> 5;
  if (doA) ASYNC_CP16(&sA [0][sltA], &A [am  * Kd + aq  * 8]);
  if (doB) ASYNC_CP16(&sBn[0][sltB], &Wn[bn2 * Kd + bq2 * 8]);

  v8f acc = {0.f,0.f,0.f,0.f,0.f,0.f,0.f,0.f};

  for (int ks = 0; ks < nT; ++ks){
    const int cur = ks & 1;
    if (ks + 1 < nT){
      const int kb = (ks + 1) * 32;                 // prefetch next tile (stays in flight)
      if (doA) ASYNC_CP16(&sA [cur ^ 1][sltA], &A [am  * Kd + kb + aq  * 8]);
      if (doB) ASYNC_CP16(&sBn[cur ^ 1][sltB], &Wn[bn2 * Kd + kb + bq2 * 8]);
      const int t = (anyA ? 1 : 0) + (anyB ? 1 : 0);
      if      (t == 2) WAIT_ASYNC(2);               // retire current tile only
      else if (t == 1) WAIT_ASYNC(1);
      else             WAIT_ASYNC(0);
    } else {
      WAIT_ASYNC(0);
    }
    __syncthreads();

    fragU fa, fb;
    fa.q[0] = *reinterpret_cast<const uint4*>(&sA [cur][baseA]);
    fa.q[1] = *reinterpret_cast<const uint4*>(&sA [cur][baseA + 16]);
    fb.q[0] = *reinterpret_cast<const uint4*>(&sBn[cur][baseB]);
    fb.q[1] = *reinterpret_cast<const uint4*>(&sBn[cur][baseB + 16]);
    acc = __builtin_amdgcn_wmma_f32_16x16x32_bf16(false, fa.v, false, fb.v,
                                                  (short)0, acc, false, false);
    __syncthreads();
  }

  const int nn = n0 + nt + lm;
#pragma unroll
  for (int r = 0; r < 8; ++r){
    int mm = m0 + mt + r + 8*lh;                       // D layout: m = r + 8*half
    if (mm < M && nn < N){
      float v = acc[r];
      if (bias)      v += bias[nn];
      if (flags & 1) v  = gelu_f(v);
      if (flags & 2){
        int b = mm / NNT, p = mm % NNT;                // scatter to (B,C,H,W)
        outF[(b * CCH + nn) * NNT + p] = v;
      } else if (outF){
        outF[mm * N + nn] = v;
      }
      if (outB) outB[mm * N + nn] = (__bf16)v;
    }
  }
}

// ---------------- K3: depthwise 3x3 stride-2 conv, pad 1, NHWC ----------------
__global__ void k_dwconv(const float* __restrict__ t1, const float* __restrict__ Wdw,
                         const float* __restrict__ bdw, float* __restrict__ t2){
  int idx = blockIdx.x * blockDim.x + threadIdx.x;
  if (idx >= BB * SSD * CCH) return;
  int c = idx % CCH;
  int p = (idx / CCH) % SSD;
  int b = idx / (CCH * SSD);
  int rh = p / RH, rw = p % RH;
  float a = bdw[c];
#pragma unroll
  for (int kh = 0; kh < 3; ++kh){
#pragma unroll
    for (int kw = 0; kw < 3; ++kw){
      int h = rh * 2 - 1 + kh, w = rw * 2 - 1 + kw;
      if (h >= 0 && h < HH && w >= 0 && w < WWD)
        a += t1[((b * NNT) + h * WWD + w) * CCH + c] * Wdw[c * 9 + kh * 3 + kw];
    }
  }
  t2[idx] = a;
}

// ---------------- K4: LayerNorm(C) + GELU -> bf16 ----------------
__global__ void k_lngelu(const float* __restrict__ t2, const float* __restrict__ g,
                         const float* __restrict__ be, __bf16* __restrict__ t3){
  __shared__ float red[128];
  const int row = blockIdx.x;                // B*49 rows
  const int tid = threadIdx.x;
  const float* p = t2 + row * CCH;

  float s = 0.f;
  for (int c = tid; c < CCH; c += 128) s += p[c];
  red[tid] = s; __syncthreads();
  for (int o = 64; o > 0; o >>= 1){ if (tid < o) red[tid] += red[tid + o]; __syncthreads(); }
  float mu = red[0] / (float)CCH; __syncthreads();

  float s2 = 0.f;
  for (int c = tid; c < CCH; c += 128){ float d = p[c] - mu; s2 += d * d; }
  red[tid] = s2; __syncthreads();
  for (int o = 64; o > 0; o >>= 1){ if (tid < o) red[tid] += red[tid + o]; __syncthreads(); }
  float rstd = rsqrtf(red[0] / (float)CCH + 1e-5f); __syncthreads();

  for (int c = tid; c < CCH; c += 128){
    float v = (p[c] - mu) * rstd * g[c] + be[c];
    t3[row * CCH + c] = (__bf16)gelu_f(v);
  }
}

// ---------------- K6: decode offsets  off[b,n,s,g] = tanh(raw)*range*FACTOR ----------------
__global__ void k_off(const float* __restrict__ oraw, float* __restrict__ offb){
  int idx = blockIdx.x * blockDim.x + threadIdx.x;
  if (idx >= BB * NNT * SSD * 2) return;
  int g = idx & 1;
  int s = (idx >> 1) % SSD;
  int n = ((idx >> 1) / SSD) % NNT;
  int b = idx / (2 * SSD * NNT);
  float raw = oraw[(b * SSD + s) * (2 * NNT) + g * NNT + n];
  offb[idx] = tanhf(raw) * (2.0f / 14.0f);   // (1/H)*FACTOR, H==W==14
}

// ---------------- K7: center offsets  co[b,n,kk,g] = tanh(q . Wco)*range ----------------
__global__ void k_co(const float* __restrict__ qf, const float* __restrict__ Wco,
                     float* __restrict__ cob){
  int idx = blockIdx.x * blockDim.x + threadIdx.x;
  if (idx >= BB * NNT * 18) return;
  int j = idx % 18, row = idx / 18;
  const float4* qr = reinterpret_cast<const float4*>(qf + row * CCH);
  const float4* wr = reinterpret_cast<const float4*>(Wco + j * CCH);
  float a = 0.f;
  for (int c4 = 0; c4 < CCH / 4; ++c4){
    float4 qa = qr[c4], wa = wr[c4];
    a += qa.x * wa.x + qa.y * wa.y + qa.z * wa.z + qa.w * wa.w;
  }
  cob[idx] = tanhf(a) * (1.0f / 14.0f);
}

// ---------------- K8: fused per-token deformable attention ----------------
// One workgroup per (b,n).  Restructured math:
//   attn[h,s] = feats[s,:] . qk[h,:] + (q[h,:] . bk[h])   (qk = q projected through Wk)
//   out[h,:]  = fbar[h,:] @ Wv^T + bv                     (fbar = attn-weighted feats)
__global__ __launch_bounds__(256) void k_attn(
    const float* __restrict__ xh,   const float* __restrict__ qf,
    const __bf16* __restrict__ WkT, const float* __restrict__ bk,
    const float* __restrict__ Wv,   const float* __restrict__ bv,
    const float* __restrict__ offb, const float* __restrict__ cob,
    const float* __restrict__ pem,  __bf16* __restrict__ obf)
{
  const int bn  = blockIdx.x;
  const int b   = bn / NNT, n = bn % NNT;
  const int hp  = n / WWD,  wp = n % WWD;
  const int tid = threadIdx.x;
  const int wv  = tid >> 5, lane = tid & 31;
  const int lm  = lane & 15, lh = lane >> 4;

  __shared__ float  sQ[CCH];                         // q token
  __shared__ __align__(16) __bf16 sQKh[16 * CCH];    // [head][c]  (B-frag: n-major)
  __shared__ float  sQKb[16];                        // q . bk per head
  __shared__ __align__(16) __bf16 sF [64 * CCH];     // [s][c]  (A-frag, stage 2)
  __shared__ __align__(16) __bf16 sFT[CCH * 64];     // [c][s]  (B-frag, stage 4)
  __shared__ float  sW[STT][4];                      // bilinear weights
  __shared__ int    sI[STT][4];                      // bilinear gather indices
  __shared__ float  sAttn[64 * 16];                  // [s][head] scores
  __shared__ __align__(16) __bf16 sAT[16 * 64];      // [head][s]  (A-frag, stage 4)
  __shared__ float  sMax[16], sSum[16];
  __shared__ __align__(16) float  sFB[16 * CCH];     // [head][c] attn-weighted feats

  // ---- stage 0a: load q ----
  for (int i = tid; i < CCH; i += 256) sQ[i] = qf[bn * CCH + i];
  __syncthreads();

  // ---- stage 0b: qk[h][c] and per-head bias term (vectorized b128 weight reads) ----
  for (int i = tid; i < 16 * CCH; i += 256){
    int hd = i / CCH, c = i - hd * CCH;
    float a = 0.f;
    if (hd < NHEAD){
      const uint4* wq = reinterpret_cast<const uint4*>(WkT + c * CCH + hd * HCD);
      W8 w0, w1, w2, w3;
      w0.q = wq[0]; w1.q = wq[1]; w2.q = wq[2]; w3.q = wq[3];
#pragma unroll
      for (int d = 0; d < 8; ++d){
        a += sQ[hd * HCD + d     ] * (float)w0.h[d];
        a += sQ[hd * HCD + d + 8 ] * (float)w1.h[d];
        a += sQ[hd * HCD + d + 16] * (float)w2.h[d];
        a += sQ[hd * HCD + d + 24] * (float)w3.h[d];
      }
    }
    sQKh[hd * CCH + c] = (__bf16)a;
  }
  if (tid < 16){
    float a = 0.f;
    if (tid < NHEAD)
      for (int d = 0; d < HCD; ++d) a += sQ[tid * HCD + d] * bk[tid * HCD + d];
    sQKb[tid] = a;
  }

  // ---- stage 1a: sampling coords (58 rows) ----
  if (tid < STT){
    float cy, cx;
    if (tid < SSD){
      int rh = tid / RH, rw = tid % RH;
      const float* o = offb + (bn * SSD + tid) * 2;
      cy = o[0] + (2.f * (float)rh) / 13.f * 2.f - 1.f;   // off + ref grid
      cx = o[1] + (2.f * (float)rw) / 13.f * 2.f - 1.f;
    } else {
      int kk = tid - SSD;
      float oy = cob[bn * 18 + kk * 2 + 0];
      float ox = cob[bn * 18 + kk * 2 + 1];
      float py = fminf(fmaxf((float)hp + OM[kk][0], 0.f), 14.f);  // clip(.,0,H)
      float px = fminf(fmaxf((float)wp + OM[kk][1], 0.f), 14.f);
      cy = oy + py / 13.f * 2.f - 1.f;
      cx = ox + px / 13.f * 2.f - 1.f;
    }
    bili(cy, cx, HH, WWD, sW[tid], sI[tid]);
  }
  __syncthreads();

  // ---- stage 1b: gather features into LDS (bf16, row + transposed copies) ----
  {
    const float* img = xh + (b * NNT) * CCH;
    for (int i = tid; i < STT * CCH; i += 256){
      int s = i / CCH, c = i - s * CCH;
      float v = sW[s][0] * img[sI[s][0] * CCH + c] + sW[s][1] * img[sI[s][1] * CCH + c]
              + sW[s][2] * img[sI[s][2] * CCH + c] + sW[s][3] * img[sI[s][3] * CCH + c];
      __bf16 bvv = (__bf16)v;
      sF [s * CCH + c] = bvv;
      sFT[c * 64 + s]  = bvv;
    }
    for (int i = tid; i < (64 - STT) * CCH; i += 256){  // zero pad rows 58..63
      int s = STT + i / CCH, c = i % CCH;
      sF [s * CCH + c] = (__bf16)0.0f;
      sFT[c * 64 + s]  = (__bf16)0.0f;
    }
  }
  __syncthreads();

  // ---- stage 2: scores = sF(64x384) x qk^T(384x16) via WMMA (waves 0..3) ----
  if (wv < 4){
    const int mt = wv * 16;
    v8f acc = {0.f,0.f,0.f,0.f,0.f,0.f,0.f,0.f};
    const int baseB0 = lm * CCH + 8 * lh;              // sQKh[n=head][k=c]
    for (int ks = 0; ks < 12; ++ks){
      const int kb = ks * 32;
      const int baseA = (mt + lm) * CCH + kb + 8 * lh;
      fragU fa, fb;
      fa.q[0] = *reinterpret_cast<const uint4*>(&sF  [baseA]);
      fa.q[1] = *reinterpret_cast<const uint4*>(&sF  [baseA + 16]);
      fb.q[0] = *reinterpret_cast<const uint4*>(&sQKh[baseB0 + kb]);
      fb.q[1] = *reinterpret_cast<const uint4*>(&sQKh[baseB0 + kb + 16]);
      acc = __builtin_amdgcn_wmma_f32_16x16x32_bf16(false, fa.v, false, fb.v,
                                                    (short)0, acc, false, false);
    }
#pragma unroll
    for (int r = 0; r < 8; ++r)
      sAttn[(mt + r + 8*lh) * 16 + lm] = acc[r];
  }
  __syncthreads();

  // ---- stage 3a: add dyn-rel-pos bias + q.bk ----
  for (int i = tid; i < STT * NHEAD; i += 256){
    int s = i / NHEAD, hd = i % NHEAD;
    float ry, rx;
    if (s < SSD){
      int rh = s / RH, rw = s % RH;
      const float* o = offb + (bn * SSD + s) * 2;
      ry = (2.f * (float)rh - (float)hp) / 13.f - o[0];   // rpp - off
      rx = (2.f * (float)rw - (float)wp) / 13.f - o[1];
    } else {
      int kk = s - SSD;
      ry = cob[bn * 18 + kk * 2 + 0] - (float)hp;         // co - center(raw)
      rx = cob[bn * 18 + kk * 2 + 1] - (float)wp;
    }
    float w4[4]; int id4[4];
    bili(ry, rx, TBD, TBD, w4, id4);
    const float* tb = pem + hd * TBD * TBD;
    float bias = w4[0]*tb[id4[0]] + w4[1]*tb[id4[1]] + w4[2]*tb[id4[2]] + w4[3]*tb[id4[3]];
    sAttn[s * 16 + hd] += bias + sQKb[hd];
  }
  __syncthreads();

  // ---- stage 3b: softmax over s (58) per head ----
  if (tid < NHEAD){
    float m = -1e30f;
    for (int s = 0; s < STT; ++s) m = fmaxf(m, sAttn[s * 16 + tid]);
    float sum = 0.f;
    for (int s = 0; s < STT; ++s) sum += __expf(sAttn[s * 16 + tid] - m);
    sMax[tid] = m; sSum[tid] = 1.f / sum;
  }
  __syncthreads();
  for (int i = tid; i < 64 * 16; i += 256){
    int s = i / 16, hd = i % 16;
    float a = 0.f;
    if (s < STT && hd < NHEAD)
      a = __expf(sAttn[i] - sMax[hd]) * sSum[hd];
    sAT[hd * 64 + s] = (__bf16)a;                      // transpose into A layout
  }
  __syncthreads();

  // ---- stage 4: fbar = sAT(16x64) x sF(64x384) via WMMA (24 N-tiles / 8 waves) ----
  for (int t = 0; t < 3; ++t){
    const int n0 = (wv * 3 + t) * 16;
    v8f acc = {0.f,0.f,0.f,0.f,0.f,0.f,0.f,0.f};
#pragma unroll
    for (int ks = 0; ks < 2; ++ks){
      const int kb = ks * 32;
      const int baseA = lm * 64 + kb + 8 * lh;         // sAT[m=head][k=s]
      const int baseB = (n0 + lm) * 64 + kb + 8 * lh;  // sFT[n=c][k=s]
      fragU fa, fb;
      fa.q[0] = *reinterpret_cast<const uint4*>(&sAT[baseA]);
      fa.q[1] = *reinterpret_cast<const uint4*>(&sAT[baseA + 16]);
      fb.q[0] = *reinterpret_cast<const uint4*>(&sFT[baseB]);
      fb.q[1] = *reinterpret_cast<const uint4*>(&sFT[baseB + 16]);
      acc = __builtin_amdgcn_wmma_f32_16x16x32_bf16(false, fa.v, false, fb.v,
                                                    (short)0, acc, false, false);
    }
#pragma unroll
    for (int r = 0; r < 8; ++r)
      sFB[(r + 8*lh) * CCH + n0 + lm] = acc[r];
  }
  __syncthreads();

  // ---- stage 5: out[o] = fbar[head(o)] . Wv[o,:] + bv[o]  (float4 row reads) ----
  for (int o = tid; o < CCH; o += 256){
    int hd = o >> 5;
    const float4* fb = reinterpret_cast<const float4*>(sFB + hd * CCH);
    const float4* wr = reinterpret_cast<const float4*>(Wv + o * CCH);
    float v = bv[o];
    for (int c4 = 0; c4 < CCH / 4; ++c4){
      float4 fa = fb[c4], wa = wr[c4];
      v += fa.x * wa.x + fa.y * wa.y + fa.z * wa.z + fa.w * wa.w;
    }
    obf[bn * CCH + o] = (__bf16)v;
  }
}

// ---------------- host launcher ----------------
extern "C" void kernel_launch(void* const* d_in, const int* in_sizes, int n_in,
                              void* d_out, int out_size, void* d_ws, size_t ws_size,
                              hipStream_t stream)
{
  (void)in_sizes; (void)n_in; (void)out_size; (void)ws_size;
  const float* x   = (const float*)d_in[0];
  const float* Wq  = (const float*)d_in[1];
  const float* bq  = (const float*)d_in[2];
  const float* Wk  = (const float*)d_in[3];
  const float* bk  = (const float*)d_in[4];
  const float* Wv  = (const float*)d_in[5];
  const float* bv  = (const float*)d_in[6];
  const float* Wo1 = (const float*)d_in[7];
  const float* bo1 = (const float*)d_in[8];
  const float* Wdw = (const float*)d_in[9];
  const float* bdw = (const float*)d_in[10];
  const float* lng = (const float*)d_in[11];
  const float* lnb = (const float*)d_in[12];
  const float* Wo2 = (const float*)d_in[13];
  const float* Wco = (const float*)d_in[14];
  const float* pem = (const float*)d_in[15];
  const float* Wp  = (const float*)d_in[16];
  const float* bp  = (const float*)d_in[17];

  char* wsp = (char*)d_ws;
  size_t off = 0;
  auto carve = [&](size_t bytes) -> void* {
    void* p = wsp + off;
    off += bytes;
    off = (off + 255) & ~(size_t)255;
    return p;
  };

  const int BN = BB * NNT;                                  // 1568 tokens
  float*  xh   = (float*) carve((size_t)BN * CCH * 4);
  __bf16* xbf  = (__bf16*)carve((size_t)BN * CCH * 2);
  __bf16* Wqb  = (__bf16*)carve((size_t)CCH * CCH * 2);     // row-major bf16 weights
  __bf16* WkT  = (__bf16*)carve((size_t)CCH * CCH * 2);     // transposed (q->Wk proj)
  __bf16* Wo1b = (__bf16*)carve((size_t)CCH * CCH * 2);
  __bf16* Wpb  = (__bf16*)carve((size_t)CCH * CCH * 2);
  __bf16* Wo2b = (__bf16*)carve((size_t)392 * CCH * 2);
  float*  qf   = (float*) carve((size_t)BN * CCH * 4);
  __bf16* qbf  = (__bf16*)carve((size_t)BN * CCH * 2);
  float*  t1   = (float*) carve((size_t)BN * CCH * 4);
  float*  t2   = (float*) carve((size_t)BB * SSD * CCH * 4);
  __bf16* t3   = (__bf16*)carve((size_t)BB * SSD * CCH * 2);
  float*  oraw = (float*) carve((size_t)BB * SSD * 392 * 4);
  float*  offb = (float*) carve((size_t)BN * SSD * 2 * 4);
  float*  cob  = (float*) carve((size_t)BN * 18 * 4);
  __bf16* obf  = (__bf16*)carve((size_t)BN * CCH * 2);

  // layout / precision prep
  k_xpose<<<(BN * CCH + 255) / 256, 256, 0, stream>>>(x, xh, xbf);
  k_cvt<<<(CCH * CCH + 255) / 256, 256, 0, stream>>>(Wq,  Wqb,  CCH * CCH);
  k_cvt<<<(CCH * CCH + 255) / 256, 256, 0, stream>>>(Wo1, Wo1b, CCH * CCH);
  k_cvt<<<(CCH * CCH + 255) / 256, 256, 0, stream>>>(Wp,  Wpb,  CCH * CCH);
  k_cvt<<<(392 * CCH + 255) / 256, 256, 0, stream>>>(Wo2, Wo2b, 392 * CCH);
  k_wt <<<(CCH * CCH + 255) / 256, 256, 0, stream>>>(Wk,  WkT,  CCH, CCH);

  // q = x @ Wq^T + bq  (f32 + bf16 copies)
  k_gemm<<<dim3((CCH + 31) / 32, (BN + 63) / 64), 256, 0, stream>>>(
      xbf, Wqb, bq, qf, qbf, BN, CCH, CCH, 0);
  // t1 = gelu(q @ Wo1^T + bo1)
  k_gemm<<<dim3((CCH + 31) / 32, (BN + 63) / 64), 256, 0, stream>>>(
      qbf, Wo1b, bo1, t1, nullptr, BN, CCH, CCH, 1);
  // depthwise conv 3x3 s2 + bias
  k_dwconv<<<(BB * SSD * CCH + 255) / 256, 256, 0, stream>>>(t1, Wdw, bdw, t2);
  // LN + gelu -> bf16
  k_lngelu<<<BB * SSD, 128, 0, stream>>>(t2, lng, lnb, t3);
  // off_raw = t3 @ Wo2^T
  k_gemm<<<dim3((392 + 31) / 32, (BB * SSD + 63) / 64), 256, 0, stream>>>(
      t3, Wo2b, nullptr, oraw, nullptr, BB * SSD, 392, CCH, 0);
  // decode offsets
  k_off<<<(BN * SSD * 2 + 255) / 256, 256, 0, stream>>>(oraw, offb);
  // center offsets
  k_co<<<(BN * 18 + 255) / 256, 256, 0, stream>>>(qf, Wco, cob);
  // fused deformable attention per token
  k_attn<<<BN, 256, 0, stream>>>(xh, qf, WkT, bk, Wv, bv, offb, cob, pem, obf);
  // final projection with NCHW scatter into d_out
  k_gemm<<<dim3((CCH + 31) / 32, (BN + 63) / 64), 256, 0, stream>>>(
      obf, Wpb, bp, (float*)d_out, nullptr, BN, CCH, CCH, 2);
}
